// VectorSharedD_55070070669892
// MI455X (gfx1250) — matrix operand order repaired and checked by reading, not compiled, not run on hardware
//
#include <hip/hip_runtime.h>

// Problem constants (match reference setup_inputs)
#define B_ 64
#define N_ 50000
#define L_ 8
#define E_ 100000
#define U_ 20000
#define G_ 20000
#define R_ 64
#define D_ 16
#define C_ 2

typedef __attribute__((ext_vector_type(16))) __bf16 v16bf;
typedef __attribute__((ext_vector_type(8)))  __bf16 v8bf;
typedef __attribute__((ext_vector_type(8)))  float  v8f;
typedef __attribute__((ext_vector_type(4)))  float  v4f;

// ---------------- zero workspace ----------------
__global__ void zero_kernel(uint4* __restrict__ p, long n16) {
    long i = (long)blockIdx.x * blockDim.x + threadIdx.x;
    long stride = (long)gridDim.x * blockDim.x;
    uint4 z = {0u, 0u, 0u, 0u};
    for (; i < n16; i += stride) p[i] = z;
}

// ---------------- pack W (L,D,D) f32 -> bf16 B-operand VGPR image ----------------
// B operand for v_wmma_f32_16x16x32_bf16 is 32(K)x16(N):
//   element t<8 of lane holds K=8*(lane/16)+t, col N = lane%16.
//   elements t>=8 hold K=16..31 -> all zero (real K is only 0..15).
__global__ void pack_w_kernel(const float* __restrict__ W, __bf16* __restrict__ Wpk) {
    int tid = threadIdx.x;            // 256 threads: (li, lane)
    int li = tid >> 5;
    int lane = tid & 31;
    __bf16* out = Wpk + ((long)li * 32 + lane) * 16;
    #pragma unroll
    for (int t = 0; t < 16; ++t) {
        float v = 0.0f;
        if (t < 8) v = W[li * (D_ * D_) + ((lane >> 4) * 8 + t) * D_ + (lane & 15)];  // W[li][k=8*(lane/16)+t][n=lane%16]
        out[t] = (__bf16)v;
    }
}

// ---------------- scatter gene embeddings into h (N,B,D) bf16 ----------------
__global__ void gene_scatter_kernel(const float* __restrict__ X,     // (B,G)
                                    const float* __restrict__ w_in,  // (D)
                                    const float* __restrict__ b_in,  // (D)
                                    const int* __restrict__ gene_map,
                                    __bf16* __restrict__ h) {
    int idx = blockIdx.x * blockDim.x + threadIdx.x;   // over G*B
    if (idx >= G_ * B_) return;
    int g = idx / B_;
    int b = idx % B_;
    float x = X[(long)b * G_ + g];
    int node = gene_map[g];
    __bf16* hp = h + (long)node * (B_ * D_) + (long)b * D_;
    v8bf lo, hi;
    #pragma unroll
    for (int d = 0; d < 8; ++d) {
        lo[d] = (__bf16)(x * w_in[d] + b_in[d]);
        hi[d] = (__bf16)(x * w_in[d + 8] + b_in[d + 8]);
    }
    *(v8bf*)(hp)     = lo;   // 16B store
    *(v8bf*)(hp + 8) = hi;   // 16B store
}

// ---------------- per-layer edge kernel: gather + WMMA + atomic scatter ----------------
__global__ __launch_bounds__(256) void edge_kernel(const __bf16* __restrict__ h,
                                                   const __bf16* __restrict__ Wpk,
                                                   const int* __restrict__ src_l,
                                                   const int* __restrict__ dst_pos_l,
                                                   float* __restrict__ agg,
                                                   int li) {
    const int lane = threadIdx.x & 31;
    const int wave = threadIdx.x >> 5;
    const int wavesTotal = gridDim.x * (blockDim.x >> 5);
    int e = blockIdx.x * (blockDim.x >> 5) + wave;

    // B operand: loaded once per wave (32 bytes/lane)
    v16bf Bop = *(const v16bf*)(Wpk + ((long)li * 32 + lane) * 16);

    const int row = lane & 15;        // M within 16-tile, also N (=d) for C scatter
    const int kh  = (lane >> 4) * 8;  // K half-offset for A, M offset for C

    for (; e < E_; e += wavesTotal) {
        int node = src_l[e];
        int up   = dst_pos_l[e];
        const __bf16* base = h + (long)node * (B_ * D_) + (row * D_ + kh);
        float* aggb = agg + (long)up * (B_ * D_) + row;

        // Issue all four A-tile loads first (one clause, incremental waits later).
        v8bf lo[4];
        #pragma unroll
        for (int m = 0; m < 4; ++m)
            lo[m] = *(const v8bf*)(base + m * (16 * D_));

        // prefetch next edge's source tile (global_prefetch_b8)
        if (e + wavesTotal < E_) {
            int nn = src_l[e + wavesTotal];
            __builtin_prefetch(h + (long)nn * (B_ * D_) + (long)lane * ((B_ * D_) / 32), 0, 1);
        }

        #pragma unroll
        for (int m = 0; m < 4; ++m) {
            // A operand: rows = batch 16m..16m+15, K = d (0..15), zero-pad K 16..31.
            v16bf A;
            #pragma unroll
            for (int i = 0; i < 8; ++i) { A[i] = lo[m][i]; A[i + 8] = (__bf16)0.0f; }

            v8f c = {};
            c = __builtin_amdgcn_wmma_f32_16x16x32_bf16(
                    false, A, false, Bop, (short)0, c, false, false);

            // C layout: VGPR k -> M = k + 8*(lane>=16), N = lane&15
            #pragma unroll
            for (int k = 0; k < 8; ++k) {
                int batch = m * 16 + k + kh;
                unsafeAtomicAdd(&aggb[(long)batch * D_], c[k]);
            }
        }
    }
}

// ---------------- per-layer update: tanh(agg + bias) -> h, reset agg ----------------
__global__ void update_kernel(float* __restrict__ agg,
                              const int* __restrict__ dst_unique_l,
                              const float* __restrict__ bias,
                              __bf16* __restrict__ h) {
    int idx = blockIdx.x * blockDim.x + threadIdx.x;  // over U*B
    if (idx >= U_ * B_) return;
    int u = idx / B_;
    int b = idx % B_;
    int node = dst_unique_l[u];
    float* ap = agg + (long)u * (B_ * D_) + (long)b * D_;
    __bf16* hp = h + (long)node * (B_ * D_) + (long)b * D_;
    const float* bp = bias + (long)node * D_;

    v4f a0 = *(const v4f*)(ap);
    v4f a1 = *(const v4f*)(ap + 4);
    v4f a2 = *(const v4f*)(ap + 8);
    v4f a3 = *(const v4f*)(ap + 12);
    v4f b0 = *(const v4f*)(bp);
    v4f b1 = *(const v4f*)(bp + 4);
    v4f b2 = *(const v4f*)(bp + 8);
    v4f b3 = *(const v4f*)(bp + 12);

    v8bf lo, hi;
    #pragma unroll
    for (int i = 0; i < 4; ++i) {
        lo[i]     = (__bf16)tanhf(a0[i] + b0[i]);
        lo[i + 4] = (__bf16)tanhf(a1[i] + b1[i]);
        hi[i]     = (__bf16)tanhf(a2[i] + b2[i]);
        hi[i + 4] = (__bf16)tanhf(a3[i] + b3[i]);
    }
    *(v8bf*)(hp)     = lo;
    *(v8bf*)(hp + 8) = hi;

    v4f z = {0.0f, 0.0f, 0.0f, 0.0f};   // reset agg (replay-safe)
    *(v4f*)(ap)      = z;
    *(v4f*)(ap + 4)  = z;
    *(v4f*)(ap + 8)  = z;
    *(v4f*)(ap + 12) = z;
}

// ---------------- head: out[b,c] = feat[b,:] . W_head[c,:] + b_head[c] ----------------
__global__ void head_kernel(const __bf16* __restrict__ h,
                            const int* __restrict__ root_ids,
                            const float* __restrict__ W_head,
                            const float* __restrict__ b_head,
                            float* __restrict__ out) {
    int tid = blockIdx.x * blockDim.x + threadIdx.x;
    if (tid >= B_ * C_) return;
    int b = tid / C_;
    int c = tid % C_;
    float acc = b_head[c];
    for (int r = 0; r < R_; ++r) {
        int node = root_ids[r];
        const __bf16* hp = h + (long)node * (B_ * D_) + (long)b * D_;
        const float* wp = W_head + (long)c * (R_ * D_) + (long)r * D_;
        #pragma unroll
        for (int d = 0; d < D_; ++d) acc += (float)hp[d] * wp[d];
    }
    out[(long)b * C_ + c] = acc;
}

extern "C" void kernel_launch(void* const* d_in, const int* in_sizes, int n_in,
                              void* d_out, int out_size, void* d_ws, size_t ws_size,
                              hipStream_t stream) {
    const float* X        = (const float*)d_in[0];   // (B,G)
    const float* w_in     = (const float*)d_in[1];   // (D)
    const float* b_in     = (const float*)d_in[2];   // (D)
    const float* W        = (const float*)d_in[3];   // (L,D,D)
    const float* bias     = (const float*)d_in[4];   // (N,D)
    const float* W_head   = (const float*)d_in[5];   // (C,R*D)
    const float* b_head   = (const float*)d_in[6];   // (C)
    const int*   gene_map = (const int*)d_in[7];     // (G)
    const int*   src      = (const int*)d_in[8];     // (L,E)
    const int*   dst_pos  = (const int*)d_in[9];     // (L,E)
    const int*   dst_uni  = (const int*)d_in[10];    // (L,U)
    const int*   root_ids = (const int*)d_in[11];    // (R)
    float* out = (float*)d_out;

    // workspace layout
    const size_t H_BYTES   = (size_t)N_ * B_ * D_ * sizeof(__bf16);   // 102.4 MB
    const size_t AGG_BYTES = (size_t)U_ * B_ * D_ * sizeof(float);    // 81.92 MB
    char* ws = (char*)d_ws;
    __bf16* h   = (__bf16*)ws;
    float*  agg = (float*)(ws + H_BYTES);
    __bf16* Wpk = (__bf16*)(ws + H_BYTES + AGG_BYTES);
    (void)ws_size; (void)in_sizes; (void)n_in; (void)out_size;

    // 1) zero h and agg
    zero_kernel<<<2048, 256, 0, stream>>>((uint4*)h, (long)(H_BYTES / 16));
    zero_kernel<<<2048, 256, 0, stream>>>((uint4*)agg, (long)(AGG_BYTES / 16));

    // 2) pack W into WMMA B-operand image
    pack_w_kernel<<<1, 256, 0, stream>>>(W, Wpk);

    // 3) scatter gene embeddings
    gene_scatter_kernel<<<(G_ * B_ + 255) / 256, 256, 0, stream>>>(X, w_in, b_in, gene_map, h);

    // 4) message-passing layers
    for (int li = 0; li < L_; ++li) {
        edge_kernel<<<4096, 256, 0, stream>>>(h, Wpk,
                                              src + (long)li * E_,
                                              dst_pos + (long)li * E_,
                                              agg, li);
        update_kernel<<<(U_ * B_ + 255) / 256, 256, 0, stream>>>(agg,
                                                                 dst_uni + (long)li * U_,
                                                                 bias, h);
    }

    // 5) head
    head_kernel<<<1, 128, 0, stream>>>(h, root_ids, W_head, b_head, out);
}
